// MinkowskiAttention_27865747817015
// MI455X (gfx1250) — matrix operand order repaired and checked by reading, not compiled, not run on hardware
//
#include <hip/hip_runtime.h>
#include <hip/hip_bf16.h>
#include <math.h>

// Problem constants (from the reference).
constexpr int Lq = 2048;   // query length
constexpr int Sk = 2048;   // key length
constexpr int NB = 2;      // batch
constexpr int EE = 1024;   // embed dim

typedef __bf16 bf16;
typedef __attribute__((ext_vector_type(16))) __bf16 v16bf;
typedef __attribute__((ext_vector_type(8)))  __bf16 bf16x8;
typedef __attribute__((ext_vector_type(8)))  float  v8f;

// ---------------------------------------------------------------------------
// fp32 -> bf16 elementwise conversion
// ---------------------------------------------------------------------------
__global__ void f32_to_bf16_kernel(const float* __restrict__ in,
                                   bf16* __restrict__ out, int n) {
  int i = blockIdx.x * blockDim.x + threadIdx.x;
  if (i < n) out[i] = (bf16)in[i];
}

// ---------------------------------------------------------------------------
// sq/sk = X @ Ws^T + bs,  X is (R, NB, E) seq-first, out is (NB, R, 3)
// ---------------------------------------------------------------------------
__global__ void spatial_proj_kernel(const float* __restrict__ X,
                                    const float* __restrict__ Ws,  // (3, E)
                                    const float* __restrict__ bs,  // (3)
                                    float* __restrict__ out,       // (NB, R, 3)
                                    int R) {
  int t = blockIdx.x * blockDim.x + threadIdx.x;
  if (t >= NB * R * 3) return;
  int c = t % 3;
  int r = (t / 3) % R;
  int n = t / (3 * R);
  const float* x = X + ((size_t)r * NB + n) * EE;
  const float* w = Ws + (size_t)c * EE;
  float acc = bs[c];
  for (int e = 0; e < EE; ++e) acc = fmaf(x[e], w[e], acc);
  out[((size_t)n * R + r) * 3 + c] = acc;
}

// ---------------------------------------------------------------------------
// Minkowski logits + mask + softmax -> bf16 weight matrix W (NB, L, S).
// One wave32 per output row; logits are cheap (3 FMAs) so recompute per pass.
// ---------------------------------------------------------------------------
__global__ __launch_bounds__(256)
void softmax_weights_kernel(const float* __restrict__ sq,   // (NB, L, 3)
                            const float* __restrict__ sk,   // (NB, S, 3)
                            const float* __restrict__ alpha,
                            const float* __restrict__ beta,
                            const float* __restrict__ tsc,
                            bf16* __restrict__ W) {         // (NB*L, S)
  const int lane = threadIdx.x & 31;
  const int wave = threadIdx.x >> 5;
  const int row  = blockIdx.x * 8 + wave;   // row in [0, NB*L)
  const int n = row / Lq;
  const int l = row % Lq;

  const float a  = alpha[0];
  const float b  = beta[0];
  const float ts = tsc[0];
  const float q0 = sq[row * 3 + 0];
  const float q1 = sq[row * 3 + 1];
  const float q2 = sq[row * 3 + 2];
  const float* skn = sk + (size_t)n * Sk * 3;
  const float NEG_INF = -__builtin_inff();

  // pass 1: row max of masked logits
  float m = NEG_INF;
  for (int s = lane; s < Sk; s += 32) {
    float dt = (float)(l - s) * ts;
    float d0 = q0 - skn[s * 3 + 0];
    float d1 = q1 - skn[s * 3 + 1];
    float d2 = q2 - skn[s * 3 + 2];
    float s2 = b * (d0 * d0 + d1 * d1 + d2 * d2) - a * dt * dt;
    float lg = (s2 > 0.f) ? NEG_INF : s2;
    m = fmaxf(m, lg);
  }
#pragma unroll
  for (int off = 16; off > 0; off >>= 1) m = fmaxf(m, __shfl_xor(m, off, 32));

  // pass 2: sum of exp
  float sum = 0.f;
  for (int s = lane; s < Sk; s += 32) {
    float dt = (float)(l - s) * ts;
    float d0 = q0 - skn[s * 3 + 0];
    float d1 = q1 - skn[s * 3 + 1];
    float d2 = q2 - skn[s * 3 + 2];
    float s2 = b * (d0 * d0 + d1 * d1 + d2 * d2) - a * dt * dt;
    float lg = (s2 > 0.f) ? NEG_INF : s2;
    sum += (lg > NEG_INF) ? __expf(lg - m) : 0.f;
  }
#pragma unroll
  for (int off = 16; off > 0; off >>= 1) sum += __shfl_xor(sum, off, 32);

  const float inv = (sum > 0.f) ? 1.f / sum : 0.f;
  bf16* Wrow = W + (size_t)row * Sk;

  // pass 3: write normalized bf16 weights
  for (int s = lane; s < Sk; s += 32) {
    float dt = (float)(l - s) * ts;
    float d0 = q0 - skn[s * 3 + 0];
    float d1 = q1 - skn[s * 3 + 1];
    float d2 = q2 - skn[s * 3 + 2];
    float s2 = b * (d0 * d0 + d1 * d1 + d2 * d2) - a * dt * dt;
    float lg = (s2 > 0.f) ? NEG_INF : s2;
    float w  = (lg > NEG_INF) ? __expf(lg - m) * inv : 0.f;
    Wrow[s] = (bf16)w;
  }
}

// ---------------------------------------------------------------------------
// WMMA bf16 GEMM (TN form): C = A @ B^T (+ bias), f32 accumulate.
//   A: (M, K) row-major bf16, row stride lda
//   B: (N, K) row-major bf16, row stride ldb  (so fragment loads are contiguous)
//   CT=false: C row-major (M, N), row stride ldc
//   CT=true : C stored transposed, (N, M) row-major, row stride ldc;
//             per-lane rows are contiguous -> one bf16x8 (b128) store per tile
// Wave tile: 2x4 grid of 16x16 WMMA tiles (32 rows x 64 cols):
//   8 v_wmma per K-step against 12 b128 loads, 8 independent accumulators.
// Block: 8 waves (2x4) -> 64 rows x 256 cols. All dims multiples of tile.
// Fragment layouts per CDNA5 ISA 7.12.2 (wave32, 16-bit operands).
// ---------------------------------------------------------------------------
template <bool CT, typename OutT>
__global__ __launch_bounds__(256)
void gemm_wmma_kernel(const bf16* __restrict__ A, int lda, long aBatch,
                      const bf16* __restrict__ B, int ldb, long bBatch,
                      OutT* __restrict__ C, int ldc, long cBatch,
                      const float* __restrict__ bias, int K) {
  const int lane = threadIdx.x & 31;
  const int wave = threadIdx.x >> 5;
  const int wm = wave >> 2;              // 0..1
  const int wn = wave & 3;               // 0..3
  const int row0 = blockIdx.y * 64 + wm * 32;    // wave: rows [row0, row0+32)
  const int col0 = blockIdx.x * 256 + wn * 64;   // wave: cols [col0, col0+64)
  const int z = blockIdx.z;
  A += (size_t)z * aBatch;
  B += (size_t)z * bBatch;
  C += (size_t)z * cBatch;

  const int grp = lane >> 4;             // lane group 0: lanes 0-15, 1: 16-31
  const int lr  = lane & 15;

  v8f acc[2][4];
#pragma unroll
  for (int mi = 0; mi < 2; ++mi)
#pragma unroll
    for (int ni = 0; ni < 4; ++ni)
      acc[mi][ni] = (v8f){0.f, 0.f, 0.f, 0.f, 0.f, 0.f, 0.f, 0.f};

  // Per-lane base pointers. A frag: lane holds row lr (per 16-row tile),
  // K chunks {grp*8 .. grp*8+7} and {16+grp*8 .. 16+grp*8+7}.
  // B frag: lane holds output column lr (a row of B), same K chunks as A.
  const bf16* Arow[2];
  Arow[0] = A + (size_t)(row0 + lr) * lda;
  Arow[1] = Arow[0] + (size_t)16 * lda;
  const bf16* Brow[4];
#pragma unroll
  for (int ni = 0; ni < 4; ++ni)
    Brow[ni] = B + (size_t)(col0 + ni * 16 + lr) * ldb;

  for (int k = 0; k < K; k += 32) {
    v16bf afrag[2];
#pragma unroll
    for (int mi = 0; mi < 2; ++mi) {
      bf16x8 lo = *(const bf16x8*)(Arow[mi] + k + grp * 8);
      bf16x8 hi = *(const bf16x8*)(Arow[mi] + k + 16 + grp * 8);
#pragma unroll
      for (int i = 0; i < 8; ++i) { afrag[mi][i] = lo[i]; afrag[mi][8 + i] = hi[i]; }
    }

    v16bf bfrag[4];
#pragma unroll
    for (int ni = 0; ni < 4; ++ni) {
      const bf16* p = Brow[ni] + k + grp * 8;
      bf16x8 b0 = *(const bf16x8*)(p);
      bf16x8 b1 = *(const bf16x8*)(p + 16);
#pragma unroll
      for (int i = 0; i < 8; ++i) { bfrag[ni][i] = b0[i]; bfrag[ni][8 + i] = b1[i]; }
    }

    if (k + 32 < K) {
      __builtin_prefetch(Arow[0] + k + 32, 0, 3);  // global_prefetch_b8
      __builtin_prefetch(Arow[1] + k + 32, 0, 3);
    }

#pragma unroll
    for (int mi = 0; mi < 2; ++mi)
#pragma unroll
      for (int ni = 0; ni < 4; ++ni)
        acc[mi][ni] = __builtin_amdgcn_wmma_f32_16x16x32_bf16(
            /*neg_a=*/false, afrag[mi], /*neg_b=*/false, bfrag[ni],
            /*c_mod=*/(short)0, acc[mi][ni], /*reuse_a=*/false, /*reuse_b=*/false);
  }

  // Epilogue. C-tile layout: VGPR j -> row (grp*8 + j), column lr (per 16x16).
  float badd[4];
#pragma unroll
  for (int ni = 0; ni < 4; ++ni)
    badd[ni] = bias ? bias[col0 + ni * 16 + lr] : 0.f;

  if constexpr (CT) {
    // Transposed store: lane writes 8 consecutive "M" positions of column
    // (col0 + ni*16 + lr) -> contiguous 16B store.
#pragma unroll
    for (int mi = 0; mi < 2; ++mi)
#pragma unroll
      for (int ni = 0; ni < 4; ++ni) {
        bf16x8 v;
#pragma unroll
        for (int j = 0; j < 8; ++j) v[j] = (bf16)(acc[mi][ni][j] + badd[ni]);
        *(bf16x8*)(C + (size_t)(col0 + ni * 16 + lr) * ldc +
                   row0 + mi * 16 + grp * 8) = v;
      }
  } else {
#pragma unroll
    for (int mi = 0; mi < 2; ++mi)
#pragma unroll
      for (int ni = 0; ni < 4; ++ni)
#pragma unroll
        for (int j = 0; j < 8; ++j)
          C[(size_t)(row0 + mi * 16 + grp * 8 + j) * ldc + col0 + ni * 16 + lr] =
              (OutT)(acc[mi][ni][j] + badd[ni]);
  }
}

// ---------------------------------------------------------------------------
// Host launcher
// ---------------------------------------------------------------------------
extern "C" void kernel_launch(void* const* d_in, const int* in_sizes, int n_in,
                              void* d_out, int out_size, void* d_ws, size_t ws_size,
                              hipStream_t stream) {
  (void)in_sizes; (void)n_in; (void)out_size; (void)ws_size;
  // setup_inputs order: query key value Wq bq Wk bk Wv bv Wo bo Ws bs time_scale alpha beta
  const float* query = (const float*)d_in[0];
  const float* key   = (const float*)d_in[1];
  const float* value = (const float*)d_in[2];
  const float* Wv    = (const float*)d_in[7];
  const float* bv    = (const float*)d_in[8];
  const float* Wo    = (const float*)d_in[9];
  const float* bo    = (const float*)d_in[10];
  const float* Ws    = (const float*)d_in[11];
  const float* bs    = (const float*)d_in[12];
  const float* tsc   = (const float*)d_in[13];
  const float* alpha = (const float*)d_in[14];
  const float* beta  = (const float*)d_in[15];
  float* out = (float*)d_out;  // (L, NB, E) fp32

  // Workspace carve-out (~44 MB total, 256B-aligned chunks).
  char* ws = (char*)d_ws;
  auto alloc = [&](size_t bytes) -> char* {
    char* p = ws;
    ws += (bytes + 255) & ~(size_t)255;
    return p;
  };
  bf16*  value_bf = (bf16*)alloc((size_t)Sk * NB * EE * sizeof(bf16)); //  8 MB
  bf16*  Wv_bf    = (bf16*)alloc((size_t)EE * EE * sizeof(bf16));      //  2 MB
  bf16*  Wo_bf    = (bf16*)alloc((size_t)EE * EE * sizeof(bf16));      //  2 MB
  bf16*  vpT      = (bf16*)alloc((size_t)NB * EE * Sk * sizeof(bf16)); //  8 MB (n,e,s)
  float* sq       = (float*)alloc((size_t)NB * Lq * 3 * sizeof(float));
  float* skp      = (float*)alloc((size_t)NB * Sk * 3 * sizeof(float));
  bf16*  Wmat     = (bf16*)alloc((size_t)NB * Lq * Sk * sizeof(bf16)); // 16 MB
  bf16*  attn_bf  = (bf16*)alloc((size_t)Lq * NB * EE * sizeof(bf16)); //  8 MB

  // 1) fp32 -> bf16 conversions
  {
    int n = Sk * NB * EE;
    f32_to_bf16_kernel<<<(n + 255) / 256, 256, 0, stream>>>(value, value_bf, n);
    n = EE * EE;
    f32_to_bf16_kernel<<<(n + 255) / 256, 256, 0, stream>>>(Wv, Wv_bf, n);
    f32_to_bf16_kernel<<<(n + 255) / 256, 256, 0, stream>>>(Wo, Wo_bf, n);
  }

  // 2) VpT[n][e][s] = (value[s,n,:] @ Wv^T + bv)[e]   (per-batch, transposed store)
  //    A = value_bf + n*E (lda = NB*E), B = Wv (N=E, K=E), C = vpT + n*E*S (ldc = S)
  gemm_wmma_kernel<true, bf16><<<dim3(EE / 256, Sk / 64, NB), 256, 0, stream>>>(
      value_bf, NB * EE, (long)EE,
      Wv_bf, EE, 0,
      vpT, Sk, (long)EE * Sk,
      bv, EE);

  // 3) sq / sk spatial projections
  {
    int n = NB * Lq * 3;
    spatial_proj_kernel<<<(n + 255) / 256, 256, 0, stream>>>(query, Ws, bs, sq, Lq);
    n = NB * Sk * 3;
    spatial_proj_kernel<<<(n + 255) / 256, 256, 0, stream>>>(key, Ws, bs, skp, Sk);
  }

  // 4) Minkowski-masked softmax -> bf16 weight matrix (NB*L, S)
  softmax_weights_kernel<<<(NB * Lq) / 8, 256, 0, stream>>>(
      sq, skp, alpha, beta, tsc, Wmat);

  // 5) attn = W @ Vp per batch, written as (L, NB, E) bf16
  //    A = Wmat_n (L, S); B = vpT_n (N=E rows, K=S) -> contiguous fragment loads;
  //    C = attn_bf + n*E (ldc = NB*E)
  gemm_wmma_kernel<false, bf16><<<dim3(EE / 256, Lq / 64, NB), 256, 0, stream>>>(
      Wmat, Sk, (long)Lq * Sk,
      vpT, Sk, (long)EE * Sk,
      attn_bf, NB * EE, (long)EE,
      nullptr, Sk);

  // 6) out = attn @ Wo^T + bo   -> fp32 (L*NB, E)
  gemm_wmma_kernel<false, float><<<dim3(EE / 256, (Lq * NB) / 64, 1), 256, 0, stream>>>(
      attn_bf, EE, 0,
      Wo_bf, EE, 0,
      out, EE, 0,
      bo, EE);
}
